// prj_module_4002909520481
// MI455X (gfx1250) — matrix operand order repaired and checked by reading, not compiled, not run on hardware
//
#include <hip/hip_runtime.h>
#include <hip/hip_bf16.h>

#define BATCH 4
#define VIEWS 180
#define DETS 256
#define IMH 256
#define IMW 256
#define PI_F 3.14159265358979323846f

// workspace layout (in float units; all region starts 32B-aligned)
#define WS_CLAMPED 0                      // 4 floats
#define WS_COS     8                      // 180
#define WS_SIN     192                    // 180
#define WS_WDET    384                    // 256
#define WS_FILT    640                    // 511
#define WS_P       1152                   // 720*256 = 184320 f32
#define WS_PF      (1152 + 184320)        // 184320 f32
#define WS_PAH     (WS_PF + 184320)       // packed A hi: 184320 bf16 = 92160 f
#define WS_PAL     (WS_PAH + 92160)       // packed A lo
#define WS_PBH     (WS_PAL + 92160)       // packed B hi: 65536 bf16 = 32768 f
#define WS_PBL     (WS_PBH + 32768)       // packed B lo

typedef __attribute__((ext_vector_type(16))) __bf16 v16bf;
typedef __attribute__((ext_vector_type(8)))  float  v8f;

// float -> bf16 round-to-nearest-even via bit manipulation; bf16 -> float by shift.
static __device__ __forceinline__ __bf16 f2bf(float x) {
  unsigned u = __builtin_bit_cast(unsigned, x);
  unsigned r = (u + 0x7FFFu + ((u >> 16) & 1u)) >> 16;
  return __builtin_bit_cast(__bf16, (unsigned short)r);
}
static __device__ __forceinline__ float bf2f(__bf16 h) {
  unsigned u = ((unsigned)__builtin_bit_cast(unsigned short, h)) << 16;
  return __builtin_bit_cast(float, u);
}

// ---------------------------------------------------------------------------
// Kernel 1: scalars + lookup tables (clamped[b], cos/sin per view, wdet, filt)
// ---------------------------------------------------------------------------
__global__ void prep_kernel(const float* __restrict__ options,
                            const float* __restrict__ gamma,
                            const float* __restrict__ beta,
                            const float* __restrict__ fv,     // (B,7)
                            const float* __restrict__ wfed,   // (1,)
                            const float* __restrict__ wp,     // (7,1)
                            const float* __restrict__ bp,     // (1,)
                            float* __restrict__ ws) {
  const int t = threadIdx.x;  // 256 threads, 1 block
  const float dDet = options[5], s2r = options[7], d2r = options[8];
  const float virdet = dDet * s2r / (s2r + d2r);

  if (t < BATCH) {
    float wgt = wfed[0] * gamma[t] + beta[t];
    float mv = bp[0];
#pragma unroll
    for (int j = 0; j < 7; ++j) mv += fv[t * 7 + j] * wp[j];
    float wcl = (wgt <= mv) ? wgt : mv;
    ws[WS_CLAMPED + t] = fmaxf(wcl, 0.0f);
  }
  if (t < VIEWS) {
    float ang = (float)t * (2.0f * PI_F / (float)VIEWS);
    ws[WS_COS + t] = cosf(ang);
    ws[WS_SIN + t] = sinf(ang);
  }
  {
    float u = ((float)t - (float)DETS * 0.5f + 0.5f) * virdet;
    ws[WS_WDET + t] = s2r / sqrtf(s2r * s2r + u * u);
  }
  for (int i = t; i < 2 * DETS - 1; i += 256) {
    int xi = i - (DETS - 1);
    float f;
    if (xi == 0)              f = 1.0f / (4.0f * virdet * virdet);
    else if ((abs(xi) & 1))   f = -1.0f / (PI_F * PI_F * (float)xi * (float)xi * virdet * virdet);
    else                      f = 0.0f;
    ws[WS_FILT + i] = f;
  }
}

// ---------------------------------------------------------------------------
// Kernel 2: forward projection with the full 256KB image staged in LDS
// (WGP has 320KB). One block per (b, view), 256 threads = detectors.
// ---------------------------------------------------------------------------
__device__ __forceinline__ float lds_sample(const float* __restrict__ im, int x, int y) {
  if ((unsigned)x >= IMW || (unsigned)y >= IMH) return 0.0f;
  return im[y * IMW + x];
}

__global__ void project_kernel(const float* __restrict__ img,
                               const float* __restrict__ proj,
                               const float* __restrict__ options,
                               const float* __restrict__ ws,
                               float* __restrict__ p) {
  extern __shared__ float s_img[];   // IMH*IMW floats = 256KB

  const int t  = threadIdx.x;         // detector index
  const int bv = blockIdx.x;          // b*VIEWS + v
  const int v  = bv % VIEWS;
  const int b  = bv / VIEWS;

  // cooperative copy: global -> LDS as float4 (b128 loads, b128 ds stores)
  {
    const float4* __restrict__ src = (const float4*)(img + (size_t)b * IMH * IMW);
    float4* dst = (float4*)s_img;
    for (int i = t; i < (IMH * IMW) / 4; i += 256) dst[i] = src[i];
  }
  __syncthreads();

  const float c  = ws[WS_COS + v];
  const float si = ws[WS_SIN + v];
  const float dDet = options[5], s2r = options[7], d2r = options[8];
  const float virdet = dDet * s2r / (s2r + d2r);
  const float coef = PI_F / (float)VIEWS;

  const float tc = (float)t - (float)(DETS - 1) * 0.5f;

  float acc = 0.0f;
  for (int s = 0; s < IMH; ++s) {
    float sv = (float)s - (float)(IMH - 1) * 0.5f;
    float xs = tc * c - sv * si;
    float ys = tc * si + sv * c;
    float ix = xs + (float)(IMW - 1) * 0.5f;
    float iy = ys + (float)(IMH - 1) * 0.5f;
    float fx0 = floorf(ix), fy0 = floorf(iy);
    int   x0 = (int)fx0,    y0 = (int)fy0;
    float fx = ix - fx0,    fy = iy - fy0;
    acc += lds_sample(s_img, x0,     y0    ) * (1.0f - fx) * (1.0f - fy)
         + lds_sample(s_img, x0 + 1, y0    ) * fx          * (1.0f - fy)
         + lds_sample(s_img, x0,     y0 + 1) * (1.0f - fx) * fy
         + lds_sample(s_img, x0 + 1, y0 + 1) * fx          * fy;
  }

  const float wdet = ws[WS_WDET + t];
  p[(size_t)bv * DETS + t] = (acc - proj[(size_t)bv * DETS + t]) * (virdet * coef) * wdet;
}

// ---------------------------------------------------------------------------
// Kernel 3a: pack GEMM operands into WMMA fragment order as bf16 hi/lo.
//   A (720x256 = p): packedA[m*256 + kb*32 + h*16 + e] = p[m*256 + kb*32 + ka(e,h)]
//     ka: A 16x32 per-lane K contiguous h*16+e (matches B)
//   B (256x256 filter matrix, F[k][d] = filt[k-d+255]):
//     packedB[nt*4096 + kb*512 + h*256 + mr*16 + e] = F[kb*32+h*16+e][nt*16+mr]
// Blocks 0..719: A rows. Blocks 720..975: B elements.
// ---------------------------------------------------------------------------
__global__ void pack_kernel(const float* __restrict__ p,
                            const float* __restrict__ ws,
                            __bf16* __restrict__ pAhi, __bf16* __restrict__ pAlo,
                            __bf16* __restrict__ pBhi, __bf16* __restrict__ pBlo) {
  const int tid = threadIdx.x;
  const int blk = blockIdx.x;
  if (blk < BATCH * VIEWS) {
    const int kb = tid >> 5, h = (tid >> 4) & 1, e = tid & 15;
    const int ka = h * 16 + e;
    float v = p[(size_t)blk * DETS + kb * 32 + ka];
    __bf16 hi = f2bf(v);
    pAhi[(size_t)blk * DETS + tid] = hi;
    pAlo[(size_t)blk * DETS + tid] = f2bf(v - bf2f(hi));
  } else {
    const float* __restrict__ filt = ws + WS_FILT;
    const int j  = (blk - BATCH * VIEWS) * 256 + tid;   // 0..65535
    const int e  = j & 15;
    const int mr = (j >> 4) & 15;
    const int h  = (j >> 8) & 1;
    const int kb = (j >> 9) & 7;
    const int nt = j >> 12;
    const int k   = kb * 32 + h * 16 + e;
    const int col = nt * 16 + mr;
    float v = filt[k - col + (DETS - 1)];
    __bf16 hi = f2bf(v);
    pBhi[j] = hi;
    pBlo[j] = f2bf(v - bf2f(hi));
  }
}

// ---------------------------------------------------------------------------
// Kernel 3b: ramp filter as GEMM, pure WMMA inner loop.
//   pf = P(720x256) x F(256x256), bf16 hi/lo split (3 WMMAs per K-step),
//   f32 accumulate. One wave per 16x16 tile; operands are contiguous,
//   32B-aligned v16bf fragments -> global_load_b128 pairs, zero VALU prep.
// ---------------------------------------------------------------------------
__global__ void filter_gemm_kernel(const __bf16* __restrict__ pAhi,
                                   const __bf16* __restrict__ pAlo,
                                   const __bf16* __restrict__ pBhi,
                                   const __bf16* __restrict__ pBlo,
                                   float* __restrict__ pf) {
  const int lane  = threadIdx.x;       // 32 threads = 1 wave
  const int nt    = blockIdx.x;        // 16 N-tiles
  const int tileM = blockIdx.y * 16;   // 45 M-tiles
  const int half  = lane >> 4;
  const int mr    = lane & 15;

  const size_t aBase = (size_t)(tileM + mr) * DETS + half * 16;
  const size_t bBase = (size_t)nt * 4096 + half * 256 + mr * 16;

  v8f c = {};
#pragma unroll
  for (int kb = 0; kb < 8; ++kb) {
    v16bf ahi = *(const v16bf*)(pAhi + aBase + kb * 32);
    v16bf alo = *(const v16bf*)(pAlo + aBase + kb * 32);
    v16bf bhi = *(const v16bf*)(pBhi + bBase + kb * 512);
    v16bf blo = *(const v16bf*)(pBlo + bBase + kb * 512);
    c = __builtin_amdgcn_wmma_f32_16x16x32_bf16(false, ahi, false, bhi, (short)0, c, false, false);
    c = __builtin_amdgcn_wmma_f32_16x16x32_bf16(false, ahi, false, blo, (short)0, c, false, false);
    c = __builtin_amdgcn_wmma_f32_16x16x32_bf16(false, alo, false, bhi, (short)0, c, false, false);
  }

#pragma unroll
  for (int r = 0; r < 8; ++r) {
    int row = tileM + half * 8 + r;    // C/D: M = half*8 + VGPR index, N = lane&15
    int col = nt * 16 + mr;
    pf[(size_t)row * DETS + col] = c[r];
  }
}

// ---------------------------------------------------------------------------
// Kernel 4: backprojection + final update. One thread per pixel; gathers are
// smooth in x so adjacent lanes hit adjacent sinogram addresses (L2-resident).
// ---------------------------------------------------------------------------
__global__ void backproject_kernel(const float* __restrict__ input,
                                   const float* __restrict__ ws,
                                   const float* __restrict__ pf,
                                   float* __restrict__ out) {
  const int x  = threadIdx.x;          // 0..255
  const int by = blockIdx.x;           // b*IMH + y
  const int y  = by % IMH;
  const int b  = by / IMH;

  const float xx = (float)x - (float)(IMW - 1) * 0.5f;
  const float yy = (float)y - (float)(IMH - 1) * 0.5f;
  const float* __restrict__ sino = pf + (size_t)b * VIEWS * DETS;

  float acc = 0.0f;
  for (int v = 0; v < VIEWS; ++v) {
    float c  = ws[WS_COS + v];
    float si = ws[WS_SIN + v];
    float tpos = xx * c + yy * si + (float)(DETS - 1) * 0.5f;
    float f0 = floorf(tpos);
    int   i0 = (int)f0;
    float fr = tpos - f0;
    float v0 = ((unsigned)i0       < DETS) ? sino[v * DETS + i0]     : 0.0f;
    float v1 = ((unsigned)(i0 + 1) < DETS) ? sino[v * DETS + i0 + 1] : 0.0f;
    acc += v0 * (1.0f - fr) + v1 * fr;
  }

  size_t idx = (size_t)b * IMH * IMW + (size_t)y * IMW + x;
  out[idx] = input[idx] - ws[WS_CLAMPED + b] * acc;
}

// ---------------------------------------------------------------------------
extern "C" void kernel_launch(void* const* d_in, const int* in_sizes, int n_in,
                              void* d_out, int out_size, void* d_ws, size_t ws_size,
                              hipStream_t stream) {
  (void)in_sizes; (void)n_in; (void)out_size; (void)ws_size;

  const float* input   = (const float*)d_in[0]; // (B,1,H,W)
  const float* proj    = (const float*)d_in[1]; // (B,1,VIEWS,DETS)
  const float* options = (const float*)d_in[2]; // (B,9) -- row 0 used
  const float* gamma   = (const float*)d_in[3];
  const float* beta    = (const float*)d_in[4];
  const float* fv      = (const float*)d_in[5];
  const float* wfed    = (const float*)d_in[6];
  const float* wp      = (const float*)d_in[7];
  const float* bp      = (const float*)d_in[8];

  float* ws  = (float*)d_ws;
  float* out = (float*)d_out;

  __bf16* pAhi = (__bf16*)(ws + WS_PAH);
  __bf16* pAlo = (__bf16*)(ws + WS_PAL);
  __bf16* pBhi = (__bf16*)(ws + WS_PBH);
  __bf16* pBlo = (__bf16*)(ws + WS_PBL);

  prep_kernel<<<1, 256, 0, stream>>>(options, gamma, beta, fv, wfed, wp, bp, ws);

  project_kernel<<<BATCH * VIEWS, 256, IMH * IMW * sizeof(float), stream>>>(
      input, proj, options, ws, ws + WS_P);

  pack_kernel<<<BATCH * VIEWS + 256, 256, 0, stream>>>(
      ws + WS_P, ws, pAhi, pAlo, pBhi, pBlo);

  dim3 ggemm(DETS / 16, (BATCH * VIEWS) / 16);   // 16 x 45 tiles
  filter_gemm_kernel<<<ggemm, 32, 0, stream>>>(pAhi, pAlo, pBhi, pBlo, ws + WS_PF);

  backproject_kernel<<<BATCH * IMH, 256, 0, stream>>>(input, ws, ws + WS_PF, out);
}
